// EvoformerIteration_62663572848702
// MI455X (gfx1250) — matrix operand order repaired and checked by reading, not compiled, not run on hardware
//
#include <hip/hip_runtime.h>
#include <hip/hip_bf16.h>
#include <math.h>

// ---------------- types / constants ----------------
typedef __bf16 v16bf __attribute__((ext_vector_type(16)));
typedef __bf16 v8bf  __attribute__((ext_vector_type(8)));
typedef __bf16 v4bf  __attribute__((ext_vector_type(4)));
typedef float  v8f   __attribute__((ext_vector_type(8)));

#define GF_BIAS    1
#define GF_RELU    2
#define GF_SIGMOID 4
#define GF_RESID   8
#define GF_PACK    16   // write qkv output packed as [(s*8+h)][row][32]

#define RR 160      // num_res
#define AA 32       // num_atoms
#define MM 4        // num_cep
#define NN 192      // R+A
#define C1 256
#define C2 128
#define ACC 32      // attention channels
#define HH 8        // heads
#define SCALE_QK 0.17677669529663687f  // 1/sqrt(32)

__device__ inline v16bf pack16(float4 x0, float4 x1, float4 y0, float4 y1) {
  v16bf a;
  a[0]=(__bf16)x0.x; a[1]=(__bf16)x0.y; a[2]=(__bf16)x0.z; a[3]=(__bf16)x0.w;
  a[4]=(__bf16)x1.x; a[5]=(__bf16)x1.y; a[6]=(__bf16)x1.z; a[7]=(__bf16)x1.w;
  a[8]=(__bf16)y0.x; a[9]=(__bf16)y0.y; a[10]=(__bf16)y0.z; a[11]=(__bf16)y0.w;
  a[12]=(__bf16)y1.x; a[13]=(__bf16)y1.y; a[14]=(__bf16)y1.z; a[15]=(__bf16)y1.w;
  return a;
}
__device__ inline v16bf ld16bf(const __bf16* lo, const __bf16* hi) {
  v8bf a = *(const v8bf*)lo;
  v8bf b = *(const v8bf*)hi;
  return __builtin_shufflevector(a, b, 0,1,2,3,4,5,6,7,8,9,10,11,12,13,14,15);
}

// ---------------- layernorm ----------------
__global__ void k_ln(const float* __restrict__ x, const float* __restrict__ w,
                     const float* __restrict__ b, float* __restrict__ y, int C) {
  int row = blockIdx.x;
  const float* xr = x + (size_t)row * C;
  __shared__ float sh[8];
  int t = threadIdx.x, nw = blockDim.x >> 5;
  float s = 0.f;
  for (int c = t; c < C; c += blockDim.x) s += xr[c];
  #pragma unroll
  for (int o = 16; o > 0; o >>= 1) s += __shfl_xor(s, o);
  if ((t & 31) == 0) sh[t >> 5] = s;
  __syncthreads();
  if (t == 0) { float a = 0; for (int i = 0; i < nw; i++) a += sh[i]; sh[0] = a; }
  __syncthreads();
  float mu = sh[0] / C;
  __syncthreads();
  float v = 0.f;
  for (int c = t; c < C; c += blockDim.x) { float d = xr[c] - mu; v += d * d; }
  #pragma unroll
  for (int o = 16; o > 0; o >>= 1) v += __shfl_xor(v, o);
  if ((t & 31) == 0) sh[t >> 5] = v;
  __syncthreads();
  if (t == 0) { float a = 0; for (int i = 0; i < nw; i++) a += sh[i]; sh[0] = a; }
  __syncthreads();
  float inv = rsqrtf(sh[0] / C + 1e-5f);
  for (int c = t; c < C; c += blockDim.x)
    y[(size_t)row * C + c] = (xr[c] - mu) * inv * w[c] + b[c];
}

// ---------------- WMMA bf16 GEMM, wave per 32x32 tile ----------------
__global__ __launch_bounds__(32)
void k_gemm(const float* __restrict__ A, const float* __restrict__ Bm,
            const float* __restrict__ bias, const float* __restrict__ resid,
            float* __restrict__ C, int M, int N, int K, int flags, float scale) {
  int row0 = blockIdx.y * 32, col0 = blockIdx.x * 32;
  int lane = threadIdx.x;
  int lmod = lane & 15, lhi = lane >> 4;
  v8f acc00 = {0.f,0.f,0.f,0.f,0.f,0.f,0.f,0.f};
  v8f acc01 = acc00, acc10 = acc00, acc11 = acc00;

  int r0 = row0 + lmod, r1 = row0 + 16 + lmod;
  int c0 = col0 + lmod, c1 = col0 + 16 + lmod;
  bool full = (row0 + 32 <= M) && (col0 + 32 <= N);

  if (full) {
    const float* A0 = A + (size_t)r0 * K;
    const float* A1 = A + (size_t)r1 * K;
    for (int k0 = 0; k0 < K; k0 += 32) {
      const float* Bp = Bm + (size_t)(k0 + 16 * lhi) * N;
      if (k0 + 32 < K) {
        __builtin_prefetch(A0 + k0 + 32 + 8 * lhi, 0, 0);
        __builtin_prefetch(Bp + (size_t)32 * N + c0, 0, 0);
      }
      v16bf a0, a1, b0, b1;
      {
        const float4* p = (const float4*)(A0 + k0 + 16 * lhi);
        const float4* q = (const float4*)(A0 + k0 + 16 * lhi + 8);
        a0 = pack16(p[0], p[1], q[0], q[1]);
      }
      {
        const float4* p = (const float4*)(A1 + k0 + 16 * lhi);
        const float4* q = (const float4*)(A1 + k0 + 16 * lhi + 8);
        a1 = pack16(p[0], p[1], q[0], q[1]);
      }
      #pragma unroll
      for (int e = 0; e < 16; e++) {
        b0[e] = (__bf16)Bp[(size_t)e * N + c0];
        b1[e] = (__bf16)Bp[(size_t)e * N + c1];
      }
      acc00 = __builtin_amdgcn_wmma_f32_16x16x32_bf16(false, a0, false, b0, (short)0, acc00, false, false);
      acc01 = __builtin_amdgcn_wmma_f32_16x16x32_bf16(false, a0, false, b1, (short)0, acc01, false, false);
      acc10 = __builtin_amdgcn_wmma_f32_16x16x32_bf16(false, a1, false, b0, (short)0, acc10, false, false);
      acc11 = __builtin_amdgcn_wmma_f32_16x16x32_bf16(false, a1, false, b1, (short)0, acc11, false, false);
    }
    if (flags & GF_PACK) {
      // col -> (s = (col%24)/8, h = col%8, c = col/24); out[((s*8+h)*M + row)*32 + c]
      int s0 = (c0 % 24) >> 3, h0 = c0 & 7, cc0 = c0 / 24;
      int s1 = (c1 % 24) >> 3, h1 = c1 & 7, cc1 = c1 / 24;
      size_t base0 = (size_t)(s0 * 8 + h0) * M * 32 + cc0;
      size_t base1 = (size_t)(s1 * 8 + h1) * M * 32 + cc1;
      #pragma unroll
      for (int r = 0; r < 8; r++) {
        int ra = row0 + r + 8 * lhi;
        int rb_ = ra + 16;
        C[base0 + (size_t)ra * 32]  = acc00[r] * scale;
        C[base1 + (size_t)ra * 32]  = acc01[r] * scale;
        C[base0 + (size_t)rb_ * 32] = acc10[r] * scale;
        C[base1 + (size_t)rb_ * 32] = acc11[r] * scale;
      }
      return;
    }
    float bs0 = (flags & GF_BIAS) ? bias[c0] : 0.f;
    float bs1 = (flags & GF_BIAS) ? bias[c1] : 0.f;
    #pragma unroll
    for (int r = 0; r < 8; r++) {
      int ra = row0 + r + 8 * lhi;
      int rb_ = ra + 16;
      float v00 = acc00[r] * scale + bs0;
      float v01 = acc01[r] * scale + bs1;
      float v10 = acc10[r] * scale + bs0;
      float v11 = acc11[r] * scale + bs1;
      if (flags & GF_RELU) {
        v00 = fmaxf(v00, 0.f); v01 = fmaxf(v01, 0.f);
        v10 = fmaxf(v10, 0.f); v11 = fmaxf(v11, 0.f);
      }
      if (flags & GF_SIGMOID) {
        v00 = 1.f/(1.f+__expf(-v00)); v01 = 1.f/(1.f+__expf(-v01));
        v10 = 1.f/(1.f+__expf(-v10)); v11 = 1.f/(1.f+__expf(-v11));
      }
      if (flags & GF_RESID) {
        v00 += resid[(size_t)ra * N + c0];  v01 += resid[(size_t)ra * N + c1];
        v10 += resid[(size_t)rb_ * N + c0]; v11 += resid[(size_t)rb_ * N + c1];
      }
      C[(size_t)ra * N + c0]  = v00; C[(size_t)ra * N + c1]  = v01;
      C[(size_t)rb_ * N + c0] = v10; C[(size_t)rb_ * N + c1] = v11;
    }
  } else {
    int r0c = r0 < M ? r0 : M - 1;
    int r1c = r1 < M ? r1 : M - 1;
    int c0c = c0 < N ? c0 : N - 1;
    int c1c = c1 < N ? c1 : N - 1;
    bool v0a = r0 < M, v1a = r1 < M, v0b = c0 < N, v1b = c1 < N;
    const float* A0 = A + (size_t)r0c * K;
    const float* A1 = A + (size_t)r1c * K;
    for (int k0 = 0; k0 < K; k0 += 32) {
      v16bf a0, a1, b0, b1;
      #pragma unroll
      for (int e = 0; e < 16; e++) {
        int kk = k0 + 16 * lhi + e;
        float f0 = A0[kk]; f0 = v0a ? f0 : 0.f;
        float f1 = A1[kk]; f1 = v1a ? f1 : 0.f;
        a0[e] = (__bf16)f0; a1[e] = (__bf16)f1;
      }
      #pragma unroll
      for (int e = 0; e < 16; e++) {
        size_t roff = (size_t)(k0 + 16 * lhi + e) * N;
        float f0 = Bm[roff + c0c]; f0 = v0b ? f0 : 0.f;
        float f1 = Bm[roff + c1c]; f1 = v1b ? f1 : 0.f;
        b0[e] = (__bf16)f0; b1[e] = (__bf16)f1;
      }
      acc00 = __builtin_amdgcn_wmma_f32_16x16x32_bf16(false, a0, false, b0, (short)0, acc00, false, false);
      acc01 = __builtin_amdgcn_wmma_f32_16x16x32_bf16(false, a0, false, b1, (short)0, acc01, false, false);
      acc10 = __builtin_amdgcn_wmma_f32_16x16x32_bf16(false, a1, false, b0, (short)0, acc10, false, false);
      acc11 = __builtin_amdgcn_wmma_f32_16x16x32_bf16(false, a1, false, b1, (short)0, acc11, false, false);
    }
    float bs0 = ((flags & GF_BIAS) && v0b) ? bias[c0] : 0.f;
    float bs1 = ((flags & GF_BIAS) && v1b) ? bias[c1] : 0.f;
    #pragma unroll
    for (int r = 0; r < 8; r++) {
      int ra = row0 + r + 8 * lhi;
      int rb_ = ra + 16;
      float vv[4] = { acc00[r] * scale + bs0, acc01[r] * scale + bs1,
                      acc10[r] * scale + bs0, acc11[r] * scale + bs1 };
      if (flags & GF_RELU)    for (int u = 0; u < 4; u++) vv[u] = fmaxf(vv[u], 0.f);
      if (flags & GF_SIGMOID) for (int u = 0; u < 4; u++) vv[u] = 1.f/(1.f+__expf(-vv[u]));
      if (ra < M && v0b)  { float z = vv[0]; if (flags & GF_RESID) z += resid[(size_t)ra * N + c0];  C[(size_t)ra * N + c0]  = z; }
      if (ra < M && v1b)  { float z = vv[1]; if (flags & GF_RESID) z += resid[(size_t)ra * N + c1];  C[(size_t)ra * N + c1]  = z; }
      if (rb_ < M && v0b) { float z = vv[2]; if (flags & GF_RESID) z += resid[(size_t)rb_ * N + c0]; C[(size_t)rb_ * N + c0] = z; }
      if (rb_ < M && v1b) { float z = vv[3]; if (flags & GF_RESID) z += resid[(size_t)rb_ * N + c1]; C[(size_t)rb_ * N + c1] = z; }
    }
  }
}

// ---------------- pair bias projection (quadrant weights) ----------------
__global__ void k_pair_bias(const float* __restrict__ pair,
                            const float* __restrict__ Wrr, const float* __restrict__ Wrl,
                            const float* __restrict__ Wlr, const float* __restrict__ Wll,
                            float* __restrict__ pb, float scale) {
  int r = blockIdx.x;
  int i = r / NN, j = r % NN;
  const float* W = (i < RR) ? (j < RR ? Wrr : Wrl) : (j < RR ? Wlr : Wll);
  const float* x = pair + (size_t)r * C2;
  int h = threadIdx.x & 7, part = threadIdx.x >> 3;
  float s = 0.f;
  for (int c = part * 32; c < part * 32 + 32; c++) s += x[c] * W[c * HH + h];
  s += __shfl_xor(s, 8);
  s += __shfl_xor(s, 16);
  if (threadIdx.x < 8) pb[(size_t)r * HH + h] = s * scale;
}

// ---------------- row attention core ----------------
__global__ __launch_bounds__(192)
void k_row_attn(const float* __restrict__ rqkv, const float* __restrict__ lqkv,
                const float* __restrict__ pb, float* __restrict__ rec_s1,
                float* __restrict__ rec_s2, float* __restrict__ lig_attn) {
  int q = blockIdx.x, h = blockIdx.y, m = blockIdx.z;
  int t = threadIdx.x;
  __shared__ float qa[32], qb[32], lg[NN], red[8];
  if (t < 32) {
    int c = t;
    if (q < RR) {
      qa[c] = rqkv[(size_t)q * 1536 + c * 48 + 24 + h];
      qb[c] = rqkv[(size_t)q * 1536 + c * 48 + 0 + h];
    } else {
      int row = m * AA + (q - RR);
      qa[c] = lqkv[(size_t)row * 1536 + c * 48 + 24 + h];
      qb[c] = lqkv[(size_t)row * 1536 + c * 48 + 0 + h];
    }
  }
  __syncthreads();
  float s = 0.f;
  if (q < RR) {
    if (t < RR) { for (int c = 0; c < 32; c++) s += qa[c] * rqkv[(size_t)t * 1536 + c * 48 + 32 + h]; }
    else { int kr = m * AA + t - RR;
           for (int c = 0; c < 32; c++) s += qb[c] * lqkv[(size_t)kr * 1536 + c * 48 + 32 + h]; }
  } else {
    if (t < RR) { for (int c = 0; c < 32; c++) s += qa[c] * rqkv[(size_t)t * 1536 + c * 48 + 8 + h]; }
    else { int kr = m * AA + t - RR;
           for (int c = 0; c < 32; c++) s += qb[c] * lqkv[(size_t)kr * 1536 + c * 48 + 8 + h]; }
  }
  s += pb[((size_t)q * NN + t) * HH + h];
  float v = s;
  #pragma unroll
  for (int o = 16; o > 0; o >>= 1) v = fmaxf(v, __shfl_xor(v, o));
  if ((t & 31) == 0) red[t >> 5] = v;
  __syncthreads();
  if (t == 0) { float mx = red[0]; for (int i = 1; i < 6; i++) mx = fmaxf(mx, red[i]); red[6] = mx; }
  __syncthreads();
  float e = __expf(s - red[6]);
  v = e;
  #pragma unroll
  for (int o = 16; o > 0; o >>= 1) v += __shfl_xor(v, o);
  if ((t & 31) == 0) red[t >> 5] = v;
  __syncthreads();
  if (t == 0) { float sm = 0; for (int i = 0; i < 6; i++) sm += red[i]; red[7] = sm; }
  __syncthreads();
  lg[t] = e / red[7];
  __syncthreads();
  if (t < 32) {
    int c = t;
    if (q < RR) {
      float s1 = 0.f, s2 = 0.f;
      for (int k = 0; k < RR; k++) s1 += lg[k] * rqkv[(size_t)k * 1536 + c * 48 + 40 + h];
      for (int k = 0; k < AA; k++) s2 += lg[RR + k] * lqkv[(size_t)(m * AA + k) * 1536 + c * 48 + 40 + h];
      if (m == 0) rec_s1[((size_t)q * 32 + c) * HH + h] = s1;
      rec_s2[(((size_t)m * RR + q) * 32 + c) * HH + h] = s2;
    } else {
      int qi = q - RR;
      float o2 = 0.f;
      for (int k = 0; k < RR; k++) o2 += lg[k] * rqkv[(size_t)k * 1536 + c * 48 + 16 + h];
      for (int k = 0; k < AA; k++) o2 += lg[RR + k] * lqkv[(size_t)(m * AA + k) * 1536 + c * 48 + 16 + h];
      lig_attn[(((size_t)m * AA + qi) * 32 + c) * HH + h] = o2;
    }
  }
}

__global__ void k_combine_rec(const float* __restrict__ s1, const float* __restrict__ s2,
                              float* __restrict__ rec_attn) {
  int i = blockIdx.x, t = threadIdx.x;
  float a = s1[(size_t)i * 256 + t];
  float b = 0.f;
  for (int m = 0; m < MM; m++) b += s2[((size_t)m * RR + i) * 256 + t];
  rec_attn[(size_t)i * 256 + t] = a + 0.25f * b;
}

// ---------------- column attention (softmax over M=4) ----------------
__global__ void k_col_attn(const float* __restrict__ qkv, float* __restrict__ outA) {
  int t = threadIdx.x;
  int i = t >> 3, h = t & 7;
  float aff[MM][MM];
  for (int a = 0; a < MM; a++) for (int b = 0; b < MM; b++) aff[a][b] = 0.f;
  for (int c = 0; c < 32; c++) {
    float qv[MM], kv[MM];
    for (int m = 0; m < MM; m++) {
      const float* base = qkv + (size_t)(m * AA + i) * 768 + c * 24;
      qv[m] = base[h]; kv[m] = base[8 + h];
    }
    for (int a = 0; a < MM; a++) for (int b = 0; b < MM; b++) aff[a][b] += qv[a] * kv[b];
  }
  float w[MM][MM];
  for (int a = 0; a < MM; a++) {
    float mx = -1e30f;
    for (int b = 0; b < MM; b++) { aff[a][b] *= SCALE_QK; mx = fmaxf(mx, aff[a][b]); }
    float sm = 0.f;
    for (int b = 0; b < MM; b++) { w[a][b] = __expf(aff[a][b] - mx); sm += w[a][b]; }
    for (int b = 0; b < MM; b++) w[a][b] /= sm;
  }
  for (int c = 0; c < 32; c++) {
    float vv[MM];
    for (int n = 0; n < MM; n++) vv[n] = qkv[(size_t)(n * AA + i) * 768 + c * 24 + 16 + h];
    for (int a = 0; a < MM; a++) {
      float o = 0.f;
      for (int n = 0; n < MM; n++) o += w[a][n] * vv[n];
      outA[(size_t)(a * AA + i) * 256 + c * HH + h] = o;
    }
  }
}

// ---------------- OPM helper kernels ----------------
__global__ void k_extract_rp(const float* __restrict__ rp, float* ri, float* rj,
                             float* li, float* lj) {
  int i = blockIdx.x, x = threadIdx.x;
  const float* p = rp + (size_t)i * 128 + x * 4;
  ri[i * 32 + x] = p[0]; rj[i * 32 + x] = p[1];
  li[i * 32 + x] = p[2]; lj[i * 32 + x] = p[3];
}
__global__ void k_extract_lp(const float* __restrict__ lp, float* ri, float* rj,
                             float* li, float* lj) {
  int r = blockIdx.x, x = threadIdx.x;
  const float* p = lp + (size_t)r * 128 + x * 4;
  ri[r * 32 + x] = p[0]; rj[r * 32 + x] = p[1];
  li[r * 32 + x] = p[2]; lj[r * 32 + x] = p[3];
}
__global__ void k_means(const float* __restrict__ l_ri, const float* __restrict__ l_rj,
                        float* lm_ri, float* lm_rj) {
  int a = blockIdx.x, x = threadIdx.x;
  float s0 = 0.f, s1 = 0.f;
  for (int m = 0; m < MM; m++) {
    s0 += l_ri[(m * AA + a) * 32 + x];
    s1 += l_rj[(m * AA + a) * 32 + x];
  }
  lm_ri[a * 32 + x] = 0.25f * s0;
  lm_rj[a * 32 + x] = 0.25f * s1;
}
__global__ void k_perm(const float* __restrict__ W, float* __restrict__ Bm) {
  int id = blockIdx.x;  // x*32+y
  int x = id >> 5, y = id & 31, o = threadIdx.x;
  Bm[(size_t)y * 4096 + x * 128 + o] = W[(size_t)id * 128 + o];
}
__global__ void k_opm_quad(const float* __restrict__ A1, const float* __restrict__ G,
                           const float* __restrict__ bias, const float* __restrict__ pairIn,
                           float* __restrict__ outPair,
                           int rb, int rn, int cb, int cn, int aFromCol, int aOff,
                           int gFromCol, int gOff, int mcount, int aMS, int gMS, float scale) {
  int bidx = blockIdx.x;
  int r_ = rb + bidx / cn, c_ = cb + bidx % cn;
  int idxA = (aFromCol ? c_ : r_) - aOff;
  int idxG = (gFromCol ? c_ : r_) - gOff;
  __shared__ float av[4 * 32];
  int o = threadIdx.x;  // 128
  if (o < mcount * 32) av[o] = A1[(o >> 5) * aMS + idxA * 32 + (o & 31)];
  __syncthreads();
  float s = 0.f;
  for (int mm = 0; mm < mcount; mm++) {
    const float* Gp = G + (size_t)mm * gMS + (size_t)idxG * 4096 + o;
    for (int x = 0; x < 32; x++) s += av[mm * 32 + x] * Gp[x * 128];
  }
  size_t pi = ((size_t)r_ * NN + c_) * C2 + o;
  outPair[pi] = pairIn[pi] + bias[o] + scale * s;
}

// ---------------- WMMA fused triangle attention ----------------
// QKVH: [(s*8+h)][36864][32] (s: 0=q,1=k,2=v); b2: [36864][8]; gate/out: [36864][256]
__global__ __launch_bounds__(128)
void k_tri_wmma(const float* __restrict__ QKVH, const float* __restrict__ b2,
                const float* __restrict__ gate, float* __restrict__ outA, int startMode) {
  int fix = blockIdx.x, h = blockIdx.y;
  int t = threadIdx.x, w = t >> 5, lane = t & 31;
  int lmod = lane & 15, lhi = lane >> 4;
  const size_t NSQ = (size_t)NN * NN;
  const float* Qb = QKVH + (size_t)(0 * 8 + h) * NSQ * 32;
  const float* Kb = QKVH + (size_t)(1 * 8 + h) * NSQ * 32;
  const float* Vb = QKVH + (size_t)(2 * 8 + h) * NSQ * 32;

  __shared__ __bf16 Ks[NN][40];    // row-major K
  __shared__ __bf16 VT[32][200];   // V transposed [c][k]
  __shared__ __bf16 Ws[4][16][200];// per-wave softmax weights, rows j-local

  // stage K and V (transposed) as bf16
  for (int idx = t; idx < NN * 8; idx += 128) {
    int row = idx >> 3, ch = (idx & 7) * 4;
    int grow = startMode ? fix * NN + row : row * NN + fix;
    float4 k4 = *(const float4*)(Kb + (size_t)grow * 32 + ch);
    v4bf kb; kb[0]=(__bf16)k4.x; kb[1]=(__bf16)k4.y; kb[2]=(__bf16)k4.z; kb[3]=(__bf16)k4.w;
    *(v4bf*)&Ks[row][ch] = kb;
    float4 v4 = *(const float4*)(Vb + (size_t)grow * 32 + ch);
    VT[ch + 0][row] = (__bf16)v4.x; VT[ch + 1][row] = (__bf16)v4.y;
    VT[ch + 2][row] = (__bf16)v4.z; VT[ch + 3][row] = (__bf16)v4.w;
  }
  __syncthreads();

  for (int jt = 0; jt < 3; jt++) {
    int j0 = (w * 3 + jt) * 16;
    int qrow = startMode ? fix * NN + (j0 + lmod) : (j0 + lmod) * NN + fix;
    const float* Qr = Qb + (size_t)qrow * 32;
    v16bf aq;
    {
      const float4* p = (const float4*)(Qr + 16 * lhi);
      const float4* q = (const float4*)(Qr + 16 * lhi + 8);
      aq = pack16(p[0], p[1], q[0], q[1]);
    }
    // logits: 12 k-tiles, one 16x16x32 WMMA each
    v8f lac[12];
    #pragma unroll
    for (int kt = 0; kt < 12; kt++) {
      v16bf bk = ld16bf(&Ks[kt * 16 + lmod][16 * lhi], &Ks[kt * 16 + lmod][16 * lhi + 8]);
      v8f z = {0.f,0.f,0.f,0.f,0.f,0.f,0.f,0.f};
      lac[kt] = __builtin_amdgcn_wmma_f32_16x16x32_bf16(false, aq, false, bk, (short)0, z, false, false);
    }
    // scale + bias
    #pragma unroll
    for (int kt = 0; kt < 12; kt++) {
      int kcol = kt * 16 + lmod;
      #pragma unroll
      for (int r = 0; r < 8; r++) {
        int j = j0 + r + 8 * lhi;
        lac[kt][r] = lac[kt][r] * SCALE_QK + b2[((size_t)j * NN + kcol) * HH + h];
      }
    }
    // row softmax (rows live in 16-lane halves)
    #pragma unroll
    for (int r = 0; r < 8; r++) {
      float mx = -1e30f;
      #pragma unroll
      for (int kt = 0; kt < 12; kt++) mx = fmaxf(mx, lac[kt][r]);
      #pragma unroll
      for (int o = 1; o < 16; o <<= 1) mx = fmaxf(mx, __shfl_xor(mx, o));
      float sm = 0.f;
      #pragma unroll
      for (int kt = 0; kt < 12; kt++) { lac[kt][r] = __expf(lac[kt][r] - mx); sm += lac[kt][r]; }
      #pragma unroll
      for (int o = 1; o < 16; o <<= 1) sm += __shfl_xor(sm, o);
      float inv = 1.f / sm;
      #pragma unroll
      for (int kt = 0; kt < 12; kt++) lac[kt][r] *= inv;
    }
    // write W to per-wave LDS (A-fragment friendly, bf16)
    #pragma unroll
    for (int kt = 0; kt < 12; kt++) {
      #pragma unroll
      for (int r = 0; r < 8; r++) {
        Ws[w][r + 8 * lhi][kt * 16 + lmod] = (__bf16)lac[kt][r];
      }
    }
    // out = W @ V : 2 c-tiles x 6 k-chunks
    #pragma unroll
    for (int ct = 0; ct < 2; ct++) {
      v8f oacc = {0.f,0.f,0.f,0.f,0.f,0.f,0.f,0.f};
      #pragma unroll
      for (int kc = 0; kc < 6; kc++) {
        v16bf aw = ld16bf(&Ws[w][lmod][kc * 32 + 16 * lhi],
                          &Ws[w][lmod][kc * 32 + 16 * lhi + 8]);
        v16bf bv = ld16bf(&VT[ct * 16 + lmod][kc * 32 + 16 * lhi],
                          &VT[ct * 16 + lmod][kc * 32 + 16 * lhi + 8]);
        oacc = __builtin_amdgcn_wmma_f32_16x16x32_bf16(false, aw, false, bv, (short)0, oacc, false, false);
      }
      int ccol = ct * 16 + lmod;
      #pragma unroll
      for (int r = 0; r < 8; r++) {
        int j = j0 + r + 8 * lhi;
        int orow = startMode ? fix * NN + j : j * NN + fix;
        size_t oi = (size_t)orow * 256 + ccol * HH + h;
        outA[oi] = oacc[r] * gate[oi];
      }
    }
  }
}

// ---------------- host orchestration ----------------
extern "C" void kernel_launch(void* const* d_in, const int* in_sizes, int n_in,
                              void* d_out, int out_size, void* d_ws, size_t ws_size,
                              hipStream_t stream) {
  (void)in_sizes; (void)n_in; (void)out_size; (void)ws_size;
  const float* rec_in  = (const float*)d_in[0];
  const float* lig_in  = (const float*)d_in[1];
  const float* pair_in = (const float*)d_in[2];
  auto P = [&](int i) { return (const float*)d_in[i]; };
  // param indices (jax pytree: sorted dict keys)
  // col:3-7  lig_trans:8-13  opm:14-29  pair_trans:30-35  rec_trans:36-41
  // row:42-55  tri_end:56-63  tri_start:64-71

  float* out_rec  = (float*)d_out;
  float* out_lig  = out_rec + RR * C1;
  float* out_pair = out_rec + RR * C1 + MM * AA * C1;

  float* ws = (float*)d_ws;
  size_t off = 0;
  auto alloc = [&](size_t n) { float* p = ws + off; off += (n + 63) & ~(size_t)63; return p; };
  float* REC_LN  = alloc(RR * C1);
  float* LIG_LN  = alloc(MM * AA * C1);
  float* RQKV    = alloc(RR * 1536);
  float* LQKV    = alloc(MM * AA * 1536);
  float* PB      = alloc((size_t)NN * NN * HH);
  float* REC_S1  = alloc(RR * 256);
  float* REC_S2  = alloc(MM * RR * 256);
  float* LIG_AT  = alloc(MM * AA * 256);
  float* REC_AT  = alloc(RR * 256);
  float* QKV3    = alloc(MM * AA * 768);
  float* COL_AT  = alloc(MM * AA * 256);
  float* T1R     = alloc(RR * 1024);
  float* T1L     = alloc(MM * AA * 1024);
  float* RP      = alloc(RR * 128);
  float* LP      = alloc(MM * AA * 128);
  float* R_RI = alloc(RR * 32); float* R_RJ = alloc(RR * 32);
  float* R_LI = alloc(RR * 32); float* R_LJ = alloc(RR * 32);
  float* L_RI = alloc(MM * AA * 32); float* L_RJ = alloc(MM * AA * 32);
  float* L_LI = alloc(MM * AA * 32); float* L_LJ = alloc(MM * AA * 32);
  float* LM_RI = alloc(AA * 32); float* LM_RJ = alloc(AA * 32);
  float* BM    = alloc(32 * 4096);
  float* G_RRq = alloc(RR * 4096);
  float* G_RLq = alloc(AA * 4096);
  float* G_LRq = alloc(AA * 4096);
  float* G_LLq = alloc(MM * AA * 4096);
  float* PLN   = alloc((size_t)NN * NN * C2);
  float* QKVH  = alloc((size_t)NN * NN * 768);   // packed tri qkv; reused as pair-trans mid
  float* PGATE = alloc((size_t)NN * NN * 256);
  float* PATTN = alloc((size_t)NN * NN * 256);
  float* PB2   = alloc((size_t)NN * NN * HH);

  auto gemm = [&](const float* A, const float* B, const float* bias, const float* resid,
                  float* C, int M, int N, int K, int flags, float scale) {
    dim3 g((N + 31) / 32, (M + 31) / 32);
    k_gemm<<<g, 32, 0, stream>>>(A, B, bias, resid, C, M, N, K, flags, scale);
  };
  auto ln = [&](const float* x, const float* w, const float* b, float* y, int rows, int C) {
    k_ln<<<rows, 128, 0, stream>>>(x, w, b, y, C);
  };

  // ===== row attention =====
  ln(rec_in, P(52), P(51), REC_LN, RR, C1);
  ln(lig_in, P(45), P(44), LIG_LN, MM * AA, C1);
  gemm(REC_LN, P(53), nullptr, nullptr, RQKV, RR, 1536, C1, 0, 1.f);
  gemm(LIG_LN, P(46), nullptr, nullptr, LQKV, MM * AA, 1536, C1, 0, 1.f);
  k_pair_bias<<<NN * NN, 32, 0, stream>>>(pair_in, P(55), P(54), P(48), P(47), PB, SCALE_QK);
  k_row_attn<<<dim3(NN, HH, MM), 192, 0, stream>>>(RQKV, LQKV, PB, REC_S1, REC_S2, LIG_AT);
  k_combine_rec<<<RR, 256, 0, stream>>>(REC_S1, REC_S2, REC_AT);
  gemm(REC_AT, P(50), P(49), rec_in, out_rec, RR, C1, 256, GF_BIAS | GF_RESID, 1.f);
  gemm(LIG_AT, P(43), P(42), lig_in, out_lig, MM * AA, C1, 256, GF_BIAS | GF_RESID, 1.f);

  // ===== column attention =====
  ln(out_lig, P(6), P(5), LIG_LN, MM * AA, C1);
  gemm(LIG_LN, P(7), nullptr, nullptr, QKV3, MM * AA, 768, C1, 0, 1.f);
  k_col_attn<<<1, 256, 0, stream>>>(QKV3, COL_AT);
  gemm(COL_AT, P(4), P(3), out_lig, out_lig, MM * AA, C1, 256, GF_BIAS | GF_RESID, 1.f);

  // ===== transitions (rec, lig) =====
  ln(out_rec, P(39), P(38), REC_LN, RR, C1);
  gemm(REC_LN, P(40), P(36), nullptr, T1R, RR, 1024, C1, GF_BIAS | GF_RELU, 1.f);
  gemm(T1R, P(41), P(37), out_rec, out_rec, RR, C1, 1024, GF_BIAS | GF_RESID, 1.f);
  ln(out_lig, P(11), P(10), LIG_LN, MM * AA, C1);
  gemm(LIG_LN, P(12), P(8), nullptr, T1L, MM * AA, 1024, C1, GF_BIAS | GF_RELU, 1.f);
  gemm(T1L, P(13), P(9), out_lig, out_lig, MM * AA, C1, 1024, GF_BIAS | GF_RESID, 1.f);

  // ===== outer product mean =====
  ln(out_rec, P(25), P(24), REC_LN, RR, C1);
  ln(out_lig, P(17), P(16), LIG_LN, MM * AA, C1);
  gemm(REC_LN, P(23), P(22), nullptr, RP, RR, 128, C1, GF_BIAS, 1.f);
  gemm(LIG_LN, P(15), P(14), nullptr, LP, MM * AA, 128, C1, GF_BIAS, 1.f);
  k_extract_rp<<<RR, 32, 0, stream>>>(RP, R_RI, R_RJ, R_LI, R_LJ);
  k_extract_lp<<<MM * AA, 32, 0, stream>>>(LP, L_RI, L_RJ, L_LI, L_LJ);
  k_means<<<AA, 32, 0, stream>>>(L_RI, L_RJ, LM_RI, LM_RJ);
  k_perm<<<1024, 128, 0, stream>>>(P(29), BM);
  gemm(R_RJ, BM, nullptr, nullptr, G_RRq, RR, 4096, 32, 0, 1.f);
  k_perm<<<1024, 128, 0, stream>>>(P(27), BM);
  gemm(LM_RJ, BM, nullptr, nullptr, G_RLq, AA, 4096, 32, 0, 1.f);
  k_perm<<<1024, 128, 0, stream>>>(P(21), BM);
  gemm(LM_RI, BM, nullptr, nullptr, G_LRq, AA, 4096, 32, 0, 1.f);
  k_perm<<<1024, 128, 0, stream>>>(P(19), BM);
  gemm(L_LJ, BM, nullptr, nullptr, G_LLq, MM * AA, 4096, 32, 0, 1.f);
  k_opm_quad<<<RR * RR, 128, 0, stream>>>(R_RI, G_RRq, P(28), pair_in, out_pair,
      0, RR, 0, RR, 0, 0, 1, 0, 1, 0, 0, 1.f);
  k_opm_quad<<<RR * AA, 128, 0, stream>>>(R_LI, G_RLq, P(26), pair_in, out_pair,
      0, RR, RR, AA, 0, 0, 1, RR, 1, 0, 0, 1.f);
  k_opm_quad<<<AA * RR, 128, 0, stream>>>(R_LJ, G_LRq, P(20), pair_in, out_pair,
      RR, AA, 0, RR, 1, 0, 0, RR, 1, 0, 0, 1.f);
  k_opm_quad<<<AA * AA, 128, 0, stream>>>(L_LI, G_LLq, P(18), pair_in, out_pair,
      RR, AA, RR, AA, 0, RR, 1, RR, 4, AA * 32, AA * 4096, 0.25f);

  // ===== triangle attention (starting) : params 64-71 =====
  ln(out_pair, P(68), P(67), PLN, NN * NN, C2);
  gemm(PLN, P(71), nullptr, nullptr, QKVH, NN * NN, 768, C2, GF_PACK, 1.f);
  gemm(PLN, P(64), nullptr, nullptr, PB2, NN * NN, 8, C2, 0, 1.f);
  gemm(PLN, P(66), P(65), nullptr, PGATE, NN * NN, 256, C2, GF_BIAS | GF_SIGMOID, 1.f);
  k_tri_wmma<<<dim3(NN, HH), 128, 0, stream>>>(QKVH, PB2, PGATE, PATTN, 1);
  gemm(PATTN, P(70), P(69), out_pair, out_pair, NN * NN, C2, 256, GF_BIAS | GF_RESID, 1.f);

  // ===== triangle attention (ending) : params 56-63 =====
  ln(out_pair, P(60), P(59), PLN, NN * NN, C2);
  gemm(PLN, P(63), nullptr, nullptr, QKVH, NN * NN, 768, C2, GF_PACK, 1.f);
  gemm(PLN, P(56), nullptr, nullptr, PB2, NN * NN, 8, C2, 0, 1.f);
  gemm(PLN, P(58), P(57), nullptr, PGATE, NN * NN, 256, C2, GF_BIAS | GF_SIGMOID, 1.f);
  k_tri_wmma<<<dim3(NN, HH), 128, 0, stream>>>(QKVH, PB2, PGATE, PATTN, 0);
  gemm(PATTN, P(62), P(61), out_pair, out_pair, NN * NN, C2, 256, GF_BIAS | GF_RESID, 1.f);

  // ===== pair transition : params 30-35 =====
  ln(out_pair, P(33), P(32), PLN, NN * NN, C2);
  gemm(PLN, P(34), P(30), nullptr, QKVH, NN * NN, 512, C2, GF_BIAS | GF_RELU, 1.f);
  gemm(QKVH, P(35), P(31), out_pair, out_pair, NN * NN, C2, 512, GF_BIAS | GF_RESID, 1.f);
}
